// HGT_63393717289653
// MI455X (gfx1250) — matrix operand order repaired and checked by reading, not compiled, not run on hardware
//
#include <hip/hip_runtime.h>
#include <hip/hip_bf16.h>
#include <math.h>

typedef __bf16 bf16_t;
typedef __attribute__((ext_vector_type(16))) __bf16 v16bf;
typedef __attribute__((ext_vector_type(8)))  float  v8f;

#define C_DIM 128
#define H_DIM 8
#define D_DIM 16

// branch-free tanh: one v_exp_f32, no exec-mask divergence (vs OCML tanhf)
__device__ __forceinline__ float tanh_fast(float y) {
    float t = __expf(-2.0f * fabsf(y));
    float r = (1.0f - t) / (1.0f + t);
    return __builtin_copysignf(r, y);
}
__device__ __forceinline__ float gelu_tanh(float x) {
    float x3 = x * x * x;
    return 0.5f * x * (1.0f + tanh_fast(0.7978845608028654f * (x + 0.044715f * x3)));
}

// ---------------------------------------------------------------------------
// Dense GEMM via WMMA bf16: Y[M, COUT] = epi( pre(A)[M,128] @ W[128,COUT] + b )
//   pre: 0=identity, 1=gelu(A)
//   epi: 0=none, 1=relu, 2=skip-blend+relu with beta = sigmoid(*skipParam)
// Block = (COUT/16) waves; each wave owns a 16-col stripe and FOUR 16-row
// subtiles (64-row block stripe): one B fragment feeds 4 WMMAs.
// Fragment layouts per CDNA5 ISA 7.12.2 (wave32).
// ---------------------------------------------------------------------------
template<int COUT>
__global__ void hgt_gemm(const float* __restrict__ A,
                         const float* __restrict__ W,
                         const float* __restrict__ bias,
                         const float* __restrict__ skipX,
                         const float* __restrict__ skipParam,
                         float* __restrict__ Y,
                         int M, int preGelu, int epi)
{
    const int lane = threadIdx.x & 31;
    const int wave = threadIdx.x >> 5;
    const int grp  = lane >> 4;       // lane half
    const int lid  = lane & 15;
    const int m0   = blockIdx.x * 64;
    const int n0   = wave * 16;

    v8f acc[4] = {{}, {}, {}, {}};

    const float* arow[4];
    #pragma unroll
    for (int t = 0; t < 4; ++t) {
        int r = m0 + t * 16 + lid;
        if (r >= M) r = M - 1;                      // clamp; masked on store
        arow[t] = A + (size_t)r * C_DIM;
    }

    #pragma unroll
    for (int kk = 0; kk < 4; ++kk) {
        // B fragment (32x16 bf16): lane holds col N=n0+lid; K = kk*32 + grp*8 + j + (j>=8 ? 8 : 0)
        const float* bcol = W + (size_t)(kk * 32 + grp * 8) * COUT + n0 + lid;
        v16bf bfrag;
        #pragma unroll
        for (int j = 0; j < 16; ++j)
            bfrag[j] = (bf16_t)bcol[(size_t)(j + (j & 8)) * COUT];

        #pragma unroll
        for (int t = 0; t < 4; ++t) {
            // A fragment (16x32 bf16): lane holds row M=lid of subtile t;
            // K runs {grp*8..+7, 16+grp*8..+7} -> four global_load_b128
            const float* ap = arow[t] + kk * 32 + grp * 8;
            float4 a0 = *(const float4*)(ap);
            float4 a1 = *(const float4*)(ap + 4);
            float4 a2 = *(const float4*)(ap + 16);
            float4 a3 = *(const float4*)(ap + 20);
            float av[16] = { a0.x, a0.y, a0.z, a0.w, a1.x, a1.y, a1.z, a1.w,
                             a2.x, a2.y, a2.z, a2.w, a3.x, a3.y, a3.z, a3.w };
            v16bf afrag;
            if (preGelu) {
                #pragma unroll
                for (int j = 0; j < 16; ++j) afrag[j] = (bf16_t)gelu_tanh(av[j]);
            } else {
                #pragma unroll
                for (int j = 0; j < 16; ++j) afrag[j] = (bf16_t)av[j];
            }
            acc[t] = __builtin_amdgcn_wmma_f32_16x16x32_bf16(
                        false, afrag, false, bfrag, (short)0, acc[t], false, false);
        }
    }

    const int   col = n0 + lid;
    const float b   = bias[col];
    float beta = 0.f;
    if (epi == 2) beta = 1.0f / (1.0f + __expf(-skipParam[0]));

    // C/D layout: acc[t][r] holds element (M = t*16 + r + 8*grp, N = col)
    #pragma unroll
    for (int t = 0; t < 4; ++t) {
        #pragma unroll
        for (int r = 0; r < 8; ++r) {
            int row = m0 + t * 16 + r + 8 * grp;
            if (row >= M) continue;
            float o = acc[t][r] + b;
            if (epi == 1) {
                o = fmaxf(o, 0.f);
            } else if (epi == 2) {
                float s = skipX[(size_t)row * C_DIM + col];
                o = fmaxf(beta * o + (1.f - beta) * s, 0.f);
            }
            Y[(size_t)row * COUT + col] = o;
        }
    }
}

// ---------------------------------------------------------------------------
// Edge phase (fp32 VALU, L2-resident gathers + atomics)
// ---------------------------------------------------------------------------

// ordered-uint encoding so atomicMax(uint) == float max; init value 0 is < all
__device__ __forceinline__ unsigned enc_f32(float f) {
    unsigned b = __float_as_uint(f);
    return (b & 0x80000000u) ? ~b : (b | 0x80000000u);
}
__device__ __forceinline__ float dec_f32(unsigned e) {
    unsigned b = (e & 0x80000000u) ? (e & 0x7FFFFFFFu) : ~e;
    return __uint_as_float(b);
}

// one thread per (edge, head): alpha = (q[dst] . (k[src] @ a_rel[h])) * p / sqrt(D)
__global__ void edge_alpha(const float* __restrict__ K, const float* __restrict__ Q,
                           const int* __restrict__ src, const int* __restrict__ dst,
                           const float* __restrict__ arel,   // [8,16,16]
                           const float* __restrict__ prel,   // [8]
                           float* __restrict__ alpha,        // [E,8]
                           unsigned* __restrict__ amaxU,     // [Nd,8]
                           int E)
{
    __shared__ float sA[H_DIM * D_DIM * D_DIM];
    __shared__ float sP[H_DIM];
    for (int i = threadIdx.x; i < H_DIM * D_DIM * D_DIM; i += blockDim.x) sA[i] = arel[i];
    if (threadIdx.x < H_DIM) sP[threadIdx.x] = prel[threadIdx.x];
    __syncthreads();

    long idx = (long)blockIdx.x * blockDim.x + threadIdx.x;
    int e = (int)(idx >> 3), h = (int)(idx & 7);
    if (e >= E) return;
    int s = src[e], d = dst[e];
    const float* kp = K + (size_t)s * C_DIM + h * D_DIM;
    const float* qp = Q + (size_t)d * C_DIM + h * D_DIM;
    const float* Am = sA + h * 256;

    float qr[16];
    #pragma unroll
    for (int f = 0; f < 16; ++f) qr[f] = qp[f];

    float acc = 0.f;
    #pragma unroll
    for (int dd = 0; dd < 16; ++dd) {
        float part = 0.f;
        #pragma unroll
        for (int f = 0; f < 16; ++f) part = fmaf(Am[dd * 16 + f], qr[f], part);
        acc = fmaf(kp[dd], part, acc);
    }
    acc *= sP[h] * 0.25f;   // 1/sqrt(16)
    alpha[(size_t)e * H_DIM + h] = acc;
    atomicMax(&amaxU[(size_t)d * H_DIM + h], enc_f32(acc));
}

// one thread per (edge, head): ex = exp(alpha - amax[dst]); den[dst] += ex
__global__ void edge_exp(const int* __restrict__ dst,
                         const unsigned* __restrict__ amaxU,
                         float* __restrict__ alpha, float* __restrict__ den, int E)
{
    long idx = (long)blockIdx.x * blockDim.x + threadIdx.x;
    int e = (int)(idx >> 3), h = (int)(idx & 7);
    if (e >= E) return;
    int d = dst[e];
    float amax = dec_f32(amaxU[(size_t)d * H_DIM + h]);
    float ex = __expf(alpha[(size_t)e * H_DIM + h] - amax);
    alpha[(size_t)e * H_DIM + h] = ex;
    atomicAdd(&den[(size_t)d * H_DIM + h], ex);
}

// one thread per (edge, channel): agg[dst,c] += (v[src,h] @ m_rel[h])[f] * ex/den
__global__ void edge_msg(const float* __restrict__ V,
                         const int* __restrict__ src, const int* __restrict__ dst,
                         const float* __restrict__ mrel,   // [8,16,16]
                         const float* __restrict__ ex,     // [E,8]
                         const float* __restrict__ den,    // [Nd,8]
                         float* __restrict__ agg, int E)
{
    __shared__ float sM[H_DIM * D_DIM * D_DIM];
    for (int i = threadIdx.x; i < H_DIM * D_DIM * D_DIM; i += blockDim.x) sM[i] = mrel[i];
    __syncthreads();

    long idx = (long)blockIdx.x * blockDim.x + threadIdx.x;
    int e = (int)(idx >> 7), c = (int)(idx & 127);
    if (e >= E) return;
    int h = c >> 4, f = c & 15;
    int s = src[e], d = dst[e];
    float w = ex[(size_t)e * H_DIM + h] / (den[(size_t)d * H_DIM + h] + 1e-16f);
    const float* vp = V + (size_t)s * C_DIM + h * D_DIM;
    const float* Mm = sM + h * 256 + f;
    float acc = 0.f;
    #pragma unroll
    for (int dd = 0; dd < 16; ++dd) acc = fmaf(vp[dd], Mm[dd * 16], acc);
    atomicAdd(&agg[(size_t)d * C_DIM + c], acc * w);
}

__global__ void fill_f32(float* p, float v, long n) {
    long i = (long)blockIdx.x * blockDim.x + threadIdx.x;
    if (i < n) p[i] = v;
}
__global__ void fill_u32(unsigned* p, unsigned v, long n) {
    long i = (long)blockIdx.x * blockDim.x + threadIdx.x;
    if (i < n) p[i] = v;
}

// ---------------------------------------------------------------------------
// Host orchestration
// ---------------------------------------------------------------------------
static inline void g128(const float* A, const float* W, const float* b,
                        const float* skipX, const float* skipP, float* Y,
                        int M, int pre, int epi, hipStream_t s) {
    hgt_gemm<128><<<dim3((M + 63) / 64), dim3(256), 0, s>>>(A, W, b, skipX, skipP, Y, M, pre, epi);
}
static inline void g64(const float* A, const float* W, const float* b, float* Y,
                       int M, hipStream_t s) {
    hgt_gemm<64><<<dim3((M + 63) / 64), dim3(128), 0, s>>>(A, W, b, nullptr, nullptr, Y, M, 0, 0);
}
static inline void fillf(float* p, float v, long n, hipStream_t s) {
    fill_f32<<<dim3((unsigned)((n + 255) / 256)), dim3(256), 0, s>>>(p, v, n);
}
static inline void fillu(unsigned* p, unsigned v, long n, hipStream_t s) {
    fill_u32<<<dim3((unsigned)((n + 255) / 256)), dim3(256), 0, s>>>(p, v, n);
}

extern "C" void kernel_launch(void* const* d_in, const int* in_sizes, int n_in,
                              void* d_out, int out_size, void* d_ws, size_t ws_size,
                              hipStream_t stream) {
    (void)in_sizes; (void)n_in; (void)out_size; (void)ws_size;

    const float* x_in[3] = { (const float*)d_in[0], (const float*)d_in[1], (const float*)d_in[2] };
    const int* esrc[4] = { (const int*)d_in[3], (const int*)d_in[5], (const int*)d_in[7], (const int*)d_in[9]  };
    const int* edst[4] = { (const int*)d_in[4], (const int*)d_in[6], (const int*)d_in[8], (const int*)d_in[10] };
    const float* lin_w = (const float*)d_in[11];
    const float* lin_b = (const float*)d_in[12];
    const float* kw = (const float*)d_in[13];
    const float* qw = (const float*)d_in[14];
    const float* vw = (const float*)d_in[15];
    const float* aw = (const float*)d_in[16];
    const float* kb = (const float*)d_in[17];
    const float* qb = (const float*)d_in[18];
    const float* vb = (const float*)d_in[19];
    const float* ab = (const float*)d_in[20];
    const float* a_rel = (const float*)d_in[21];
    const float* m_rel = (const float*)d_in[22];
    const float* p_rel = (const float*)d_in[23];
    const float* skip  = (const float*)d_in[24];
    const float* out_w = (const float*)d_in[25];
    const float* out_b = (const float*)d_in[26];

    static const int NNh[3]  = { 30000, 60000, 6000 };
    static const int NEh[4]  = { 200000, 300000, 200000, 100000 };
    static const int ETSs[4] = { 0, 1, 1, 2 };
    static const int ETSd[4] = { 1, 1, 2, 0 };

    const size_t NT = (size_t)96000 * C_DIM;                 // 12,288,000 floats
    const size_t xoff[3] = { 0, (size_t)30000 * C_DIM, (size_t)90000 * C_DIM };

    float* ws    = (float*)d_ws;
    float* XS    = ws;               // node features (updated per layer)
    float* Kbuf  = ws + 1 * NT;
    float* Qbuf  = ws + 2 * NT;
    float* Vbuf  = ws + 3 * NT;
    float* AG    = ws + 4 * NT;
    float* ALPHA = ws + 5 * NT;                         // up to 300000*8
    float* DEN   = ALPHA + (size_t)300000 * H_DIM;      // up to 60000*8
    unsigned* AMAX = (unsigned*)(DEN + (size_t)60000 * H_DIM);

    // Input linear + relu per node type
    for (int i = 0; i < 3; ++i)
        g128(x_in[i], lin_w + (size_t)i * 16384, lin_b + (size_t)i * C_DIM,
             nullptr, nullptr, XS + xoff[i], NNh[i], 0, 1, stream);

    const int L = 2;
    for (int l = 0; l < L; ++l) {
        // K/Q/V projections (WMMA GEMMs)
        for (int i = 0; i < 3; ++i) {
            size_t wof = (size_t)(l * 3 + i) * 16384;
            size_t bof = (size_t)(l * 3 + i) * C_DIM;
            g128(XS + xoff[i], kw + wof, kb + bof, nullptr, nullptr, Kbuf + xoff[i], NNh[i], 0, 0, stream);
            g128(XS + xoff[i], qw + wof, qb + bof, nullptr, nullptr, Qbuf + xoff[i], NNh[i], 0, 0, stream);
            g128(XS + xoff[i], vw + wof, vb + bof, nullptr, nullptr, Vbuf + xoff[i], NNh[i], 0, 0, stream);
        }
        fillf(AG, 0.f, (long)NT, stream);

        // Edge types, sequentially (softmax is per edge type)
        for (int et = 0; et < 4; ++et) {
            int si = ETSs[et], di = ETSd[et];
            int E = NEh[et], Nd = NNh[di];
            const float* ar = a_rel + (size_t)(l * 4 + et) * 2048;
            const float* mr = m_rel + (size_t)(l * 4 + et) * 2048;
            const float* pr = p_rel + (size_t)(l * 4 + et) * H_DIM;

            fillu(AMAX, 0u, (long)Nd * H_DIM, stream);
            fillf(DEN, 0.f, (long)Nd * H_DIM, stream);

            long t1 = (long)E * H_DIM;
            edge_alpha<<<dim3((unsigned)((t1 + 255) / 256)), dim3(256), 0, stream>>>(
                Kbuf + xoff[si], Qbuf + xoff[di], esrc[et], edst[et], ar, pr, ALPHA, AMAX, E);
            edge_exp<<<dim3((unsigned)((t1 + 255) / 256)), dim3(256), 0, stream>>>(
                edst[et], AMAX, ALPHA, DEN, E);
            long t3 = (long)E * C_DIM;
            edge_msg<<<dim3((unsigned)((t3 + 255) / 256)), dim3(256), 0, stream>>>(
                Vbuf + xoff[si], esrc[et], edst[et], mr, ALPHA, DEN, AG + xoff[di], E);
        }

        // o = gelu(agg) @ aw + ab ; xs = relu(beta*o + (1-beta)*xs)   (in place)
        for (int i = 0; i < 3; ++i) {
            size_t wof = (size_t)(l * 3 + i) * 16384;
            size_t bof = (size_t)(l * 3 + i) * C_DIM;
            g128(AG + xoff[i], aw + wof, ab + bof,
                 XS + xoff[i], skip + (size_t)(l * 3 + i), XS + xoff[i],
                 NNh[i], /*preGelu=*/1, /*epi=*/2, stream);
        }
    }

    // Output projection, concatenated flat (poi, road, region)
    float* out = (float*)d_out;
    const size_t ooff[3] = { 0, (size_t)30000 * 64, (size_t)90000 * 64 };
    for (int i = 0; i < 3; ++i)
        g64(XS + xoff[i], out_w, out_b, out + ooff[i], NNh[i], stream);
}
